// RNNsearch_36567351558879
// MI455X (gfx1250) — compile-verified, matrix operand order checked by reading, not run on hardware
//
#include <hip/hip_runtime.h>

// ---------------------------------------------------------------------------
// RNNsearch (Bahdanau attention seq2seq) forward pass for gfx1250 (MI455X).
// All GEMMs via v_wmma_f32_16x16x32_bf16 (wave32 WMMA), weights converted once
// per launch to zero-padded bf16 and kept L2-resident (~90MB << 192MB L2).
// ---------------------------------------------------------------------------

typedef __bf16 bf16_t;
typedef __attribute__((ext_vector_type(16))) __bf16 v16bf;
typedef __attribute__((ext_vector_type(8)))  float  v8f;

#define BB    64      // batch
#define SS    50      // src/tgt seq len
#define E_DIM 620
#define E_PAD 640
#define H_DIM 1000
#define H_PAD 1024
#define C2DIM 2000
#define C2PAD 2048
#define M_DIM 500
#define M_PAD 512
#define M2PAD 1024    // 2*M padded
#define V_DIM 30000
#define V_PAD 30016

// ---------------------------------------------------------------------------
// Generic WMMA GEMM:  C[m,n] = sum_k A[m,k] * W[n,k] (+ bias[n])
//   A : [M, lda] bf16 row-major, M % 32 == 0, lda % 4 == 0
//   W : [n_pad, ldw] bf16 row-major (weight layout [out,in], zero padded)
//   kdim % 32 == 0, ldw % 16 == 0, n_pad % 64 == 0
//   Writes fp32 C (row stride ldc) and/or bf16 Cbf; cols >= n_valid skipped.
// Block: 256 thr = 8 waves = 2 M-subtiles x 4 N-groups -> 32 x 256 block tile.
// Inner loop is branch-free: out-of-range N tiles are clamped (duplicate
// compute, stores suppressed) so WMMA always executes with EXEC all-ones.
// ---------------------------------------------------------------------------
__global__ __launch_bounds__(256)
void gemm_bf16_wmma(const bf16_t* __restrict__ A, const bf16_t* __restrict__ W,
                    float* __restrict__ C, bf16_t* __restrict__ Cbf,
                    const float* __restrict__ bias,
                    int kdim, int lda, int ldw, long long ldc,
                    int n_pad, int n_valid)
{
  __shared__ alignas(32) bf16_t As[32 * 32];   // 2KB A tile

  const int tid   = threadIdx.x;
  const int wave  = tid >> 5;
  const int lane  = tid & 31;
  const int wm    = wave & 1;                  // M subtile within block
  const int wn    = wave >> 1;                 // N group of 64
  const int mblk  = blockIdx.y * 32;
  const int nblk  = blockIdx.x * 256 + wn * 64;
  const int lrow  = lane & 15;
  const int lhalf = lane >> 4;
  const int nlim  = n_pad - 16;

  // Clamped (always in-bounds) B row pointers for the 4 N subtiles.
  const bf16_t* wrow[4];
#pragma unroll
  for (int j = 0; j < 4; ++j) {
    const int nb = min(nblk + j * 16, nlim);
    wrow[j] = W + (long long)(nb + lrow) * ldw + lhalf * 16;
  }
  const bf16_t* arow = A + (long long)(mblk + (tid >> 3)) * lda + (tid & 7) * 4;

  v8f acc[4] = {};

  for (int k0 = 0; k0 < kdim; k0 += 32) {
    // stage 32x32 A tile: each thread moves 4 bf16 (8 bytes)
    *(uint2*)&As[(tid >> 3) * 32 + (tid & 7) * 4] = *(const uint2*)(arow + k0);
    __syncthreads();

    // A fragment (16x32, contiguous k order, same as B): row = lane%16, K-half = lane/16
    v16bf a = *(const v16bf*)&As[(wm * 16 + lrow) * 32 + lhalf * 16];

#pragma unroll
    for (int j = 0; j < 4; ++j) {
      const bf16_t* wp = wrow[j] + k0;
      v16bf b = *(const v16bf*)wp;
      __builtin_prefetch(wp + 32, 0, 0);       // speculative; OOB dropped by HW
      acc[j] = __builtin_amdgcn_wmma_f32_16x16x32_bf16(
          false, a, false, b, (short)0, acc[j], false, false);
    }
    __syncthreads();
  }

  // epilogue: lane n = lane%16; VGPR r covers M = r + 8*(lane/16)
  const int mbase = mblk + wm * 16 + lhalf * 8;
#pragma unroll
  for (int j = 0; j < 4; ++j) {
    const int nb = nblk + j * 16 + lrow;       // un-clamped: guards dup tiles
    const float bv = (bias && nb < n_pad) ? bias[nb] : 0.0f;
#pragma unroll
    for (int r = 0; r < 8; ++r) {
      if (nb < n_valid) {
        const float v = acc[j][r] + bv;
        const long long o = (long long)(mbase + r) * ldc + nb;
        if (C)   C[o]   = v;
        if (Cbf) Cbf[o] = (bf16_t)v;
      }
    }
  }
}

// ---------------------------------------------------------------------------
// Weight / bias packing (fp32 -> zero-padded bf16).  K-split remaps the
// concat-context dimension: cols [0,split) -> k, [split_pad,..) -> k-split_pad+split.
// For plain matrices pass split=K, split_pad=K_pad.
// ---------------------------------------------------------------------------
__global__ __launch_bounds__(256)
void pack_w_kernel(const float* __restrict__ src, bf16_t* __restrict__ dst,
                   int N, int K, int K_pad, int split, int split_pad)
{
  const long long idx = (long long)blockIdx.x * 256 + threadIdx.x;
  const int r  = (int)(idx / K_pad);
  const int cp = (int)(idx % K_pad);
  float v = 0.0f;
  if (r < N) {
    if (cp < split)            v = src[(long long)r * K + cp];
    else if (cp >= split_pad) {
      const int k = cp - split_pad + split;
      if (k < K)               v = src[(long long)r * K + k];
    }
  }
  dst[idx] = (bf16_t)v;
}

__global__ __launch_bounds__(256)
void pack_b_kernel(const float* __restrict__ src, float* __restrict__ dst, int N)
{
  const int idx = blockIdx.x * 256 + threadIdx.x;
  dst[idx] = (idx < N) ? src[idx] : 0.0f;
}

__global__ __launch_bounds__(256)
void zero_f32_kernel(float* p)  { p[blockIdx.x * 256 + threadIdx.x] = 0.0f; }
__global__ __launch_bounds__(256)
void zero_bf16_kernel(bf16_t* p){ p[blockIdx.x * 256 + threadIdx.x] = (bf16_t)0.0f; }

// ---------------------------------------------------------------------------
// Embedding gathers (time-major [S, B, E_PAD], bf16)
// ---------------------------------------------------------------------------
__global__ __launch_bounds__(256)
void gather_src_kernel(const float* __restrict__ emb, const int* __restrict__ tok,
                       bf16_t* __restrict__ dst)
{
  const int idx = blockIdx.x * 256 + threadIdx.x;     // SS*BB*E_PAD
  const int row = idx / E_PAD, e = idx % E_PAD;
  const int sp = row / BB, b = row % BB;
  float v = 0.0f;
  if (e < E_DIM) {
    const int t = tok[b * SS + sp];
    v = emb[(long long)t * E_DIM + e];
  }
  dst[idx] = (bf16_t)v;
}

__global__ __launch_bounds__(256)
void gather_tgt_kernel(const float* __restrict__ emb, const int* __restrict__ tok,
                       bf16_t* __restrict__ dst)   // y_prev: shifted right by 1
{
  const int idx = blockIdx.x * 256 + threadIdx.x;
  const int row = idx / E_PAD, e = idx % E_PAD;
  const int sp = row / BB, b = row % BB;
  float v = 0.0f;
  if (sp > 0 && e < E_DIM) {
    const int t = tok[b * SS + sp - 1];
    v = emb[(long long)t * E_DIM + e];
  }
  dst[idx] = (bf16_t)v;
}

// ---------------------------------------------------------------------------
// GRU elementwise phases.  X3 pre-offset to timestep row block [BB, 3*H_PAD]
// packed [r | z | cand]; Gc3 (decoder context projections) nullable.
// ---------------------------------------------------------------------------
__device__ __forceinline__ float sigmf(float x) { return 1.0f / (1.0f + __expf(-x)); }

__global__ __launch_bounds__(256)
void gru_rz_kernel(const bf16_t* __restrict__ X3, const float* __restrict__ G2,
                   const float* __restrict__ Gc3, const float* __restrict__ s,
                   const float* __restrict__ bias3,
                   bf16_t* __restrict__ rs, float* __restrict__ z)
{
  const int idx = blockIdx.x * 256 + threadIdx.x;     // BB*H_PAD
  const int b = idx >> 10, h = idx & (H_PAD - 1);
  const float xr = (float)X3[(long long)b * (3 * H_PAD) + h];
  const float xz = (float)X3[(long long)b * (3 * H_PAD) + H_PAD + h];
  const float cr = Gc3 ? Gc3[b * (3 * H_PAD) + h] : 0.0f;
  const float cz = Gc3 ? Gc3[b * (3 * H_PAD) + H_PAD + h] : 0.0f;
  const float r  = sigmf(xr + G2[b * (2 * H_PAD) + h] + cr + bias3[h]);
  const float zz = sigmf(xz + G2[b * (2 * H_PAD) + H_PAD + h] + cz + bias3[H_PAD + h]);
  rs[idx] = (bf16_t)(r * s[idx]);
  z[idx]  = zz;
}

__global__ __launch_bounds__(256)
void gru_upd_kernel(const bf16_t* __restrict__ X3, const float* __restrict__ Gc,
                    const float* __restrict__ Gc3, const float* __restrict__ z,
                    const float* __restrict__ bias3,
                    float* __restrict__ s, bf16_t* __restrict__ sbf,
                    bf16_t* __restrict__ ann)
{
  const int idx = blockIdx.x * 256 + threadIdx.x;
  const int b = idx >> 10, h = idx & (H_PAD - 1);
  const float xc = (float)X3[(long long)b * (3 * H_PAD) + 2 * H_PAD + h];
  const float cc = Gc3 ? Gc3[b * (3 * H_PAD) + 2 * H_PAD + h] : 0.0f;
  const float cand = tanhf(xc + Gc[idx] + cc + bias3[2 * H_PAD + h]);
  const float zz = z[idx];
  const float sn = (1.0f - zz) * s[idx] + zz * cand;
  s[idx]   = sn;
  sbf[idx] = (bf16_t)sn;
  if (ann) ann[idx] = (bf16_t)sn;
}

// Annotations [b*S+s, C2PAD]: hf at cols [0,H_PAD), hb at [H_PAD, 2*H_PAD)
__global__ __launch_bounds__(256)
void hann_kernel(const bf16_t* __restrict__ hf, const bf16_t* __restrict__ hb,
                 bf16_t* __restrict__ Hann)
{
  const int idx = blockIdx.x * 256 + threadIdx.x;     // BB*SS*C2PAD
  const int row = idx >> 11, j = idx & (C2PAD - 1);
  const int b = row / SS, sp = row % SS;
  bf16_t v;
  if (j < H_PAD) v = hf[(long long)(sp * BB + b) * H_PAD + j];
  else           v = hb[(long long)(sp * BB + b) * H_PAD + (j - H_PAD)];
  Hann[idx] = v;
}

// ---------------------------------------------------------------------------
// Attention: e[b,s] = v_a . tanh(W_a s + U_a h)   (Was, Uh precomputed)
// ---------------------------------------------------------------------------
__global__ __launch_bounds__(128)
void energy_kernel(const float* __restrict__ Was, const float* __restrict__ Uh,
                   const float* __restrict__ va, float* __restrict__ e)
{
  const int bs = blockIdx.x;            // b*SS + s
  const int b  = bs / SS;
  __shared__ float red[128];
  float acc = 0.0f;
  for (int h = threadIdx.x; h < H_PAD; h += 128)
    acc += tanhf(Was[b * H_PAD + h] + Uh[(long long)bs * H_PAD + h]) * va[h];
  red[threadIdx.x] = acc;
  __syncthreads();
  for (int o = 64; o > 0; o >>= 1) {
    if ((int)threadIdx.x < o) red[threadIdx.x] += red[threadIdx.x + o];
    __syncthreads();
  }
  if (threadIdx.x == 0) e[bs] = red[0];
}

__global__ __launch_bounds__(64)
void softmax_kernel(const float* __restrict__ e, float* __restrict__ alpha)
{
  const int b = blockIdx.x, i = threadIdx.x;          // 64 threads, S=50
  __shared__ float sh[64];
  const float v = (i < SS) ? e[b * SS + i] : -3.0e38f;
  sh[i] = v; __syncthreads();
  for (int o = 32; o > 0; o >>= 1) { if (i < o) sh[i] = fmaxf(sh[i], sh[i + o]); __syncthreads(); }
  const float mx = sh[0]; __syncthreads();
  const float ex = (i < SS) ? __expf(v - mx) : 0.0f;
  sh[i] = ex; __syncthreads();
  for (int o = 32; o > 0; o >>= 1) { if (i < o) sh[i] += sh[i + o]; __syncthreads(); }
  if (i < SS) alpha[b * SS + i] = ex / sh[0];
}

__global__ __launch_bounds__(256)
void context_kernel(const float* __restrict__ alpha, const bf16_t* __restrict__ Hann,
                    bf16_t* __restrict__ c)
{
  const int idx = blockIdx.x * 256 + threadIdx.x;     // BB*C2PAD
  const int b = idx >> 11, j = idx & (C2PAD - 1);
  const bf16_t* hp = Hann + (long long)b * SS * C2PAD + j;
  const float*  ap = alpha + b * SS;
  float acc = 0.0f;
#pragma unroll 5
  for (int sp = 0; sp < SS; ++sp) acc += ap[sp] * (float)hp[(long long)sp * C2PAD];
  c[idx] = (bf16_t)acc;
}

// maxout: t = U_o s + V_o y + C_o c ; tmax[m] = max(t[2m], t[2m+1])
__global__ __launch_bounds__(256)
void maxout_kernel(const float* __restrict__ T1, const float* __restrict__ T2,
                   const bf16_t* __restrict__ Yv, bf16_t* __restrict__ tmax)
{
  const int idx = blockIdx.x * 256 + threadIdx.x;     // BB*M_PAD
  const int b = idx >> 9, mi = idx & (M_PAD - 1);
  float v = 0.0f;
  if (mi < M_DIM) {
    const int j0 = 2 * mi, j1 = j0 + 1;
    const float t0 = T1[b * M2PAD + j0] + T2[b * M2PAD + j0] + (float)Yv[(long long)b * M2PAD + j0];
    const float t1 = T1[b * M2PAD + j1] + T2[b * M2PAD + j1] + (float)Yv[(long long)b * M2PAD + j1];
    v = fmaxf(t0, t1);
  }
  tmax[idx] = (bf16_t)v;
}

__global__ __launch_bounds__(256)
void tanh_s0_kernel(const float* __restrict__ in, float* __restrict__ s,
                    bf16_t* __restrict__ sbf)
{
  const int idx = blockIdx.x * 256 + threadIdx.x;     // BB*H_PAD
  const float v = tanhf(in[idx]);
  s[idx] = v; sbf[idx] = (bf16_t)v;
}

// ---------------------------------------------------------------------------
// Host orchestration
// ---------------------------------------------------------------------------
extern "C" void kernel_launch(void* const* d_in, const int* in_sizes, int n_in,
                              void* d_out, int out_size, void* d_ws, size_t ws_size,
                              hipStream_t stream)
{
  (void)in_sizes; (void)n_in; (void)out_size; (void)ws_size;

  const int*   src_tok = (const int*)d_in[0];
  const int*   tgt_tok = (const int*)d_in[1];
  const float* src_emb = (const float*)d_in[2];
  const float* tgt_emb = (const float*)d_in[3];
  // _gru_params order: W, Wz, Wr, U, Uz, Ur, b, bz, br [, C, Cz, Cr]
  const float* efW  = (const float*)d_in[4];  const float* efWz = (const float*)d_in[5];
  const float* efWr = (const float*)d_in[6];  const float* efU  = (const float*)d_in[7];
  const float* efUz = (const float*)d_in[8];  const float* efUr = (const float*)d_in[9];
  const float* efb  = (const float*)d_in[10]; const float* efbz = (const float*)d_in[11];
  const float* efbr = (const float*)d_in[12];
  const float* ebW  = (const float*)d_in[13]; const float* ebWz = (const float*)d_in[14];
  const float* ebWr = (const float*)d_in[15]; const float* ebU  = (const float*)d_in[16];
  const float* ebUz = (const float*)d_in[17]; const float* ebUr = (const float*)d_in[18];
  const float* ebb  = (const float*)d_in[19]; const float* ebbz = (const float*)d_in[20];
  const float* ebbr = (const float*)d_in[21];
  const float* dW   = (const float*)d_in[22]; const float* dWz  = (const float*)d_in[23];
  const float* dWr  = (const float*)d_in[24]; const float* dU   = (const float*)d_in[25];
  const float* dUz  = (const float*)d_in[26]; const float* dUr  = (const float*)d_in[27];
  const float* db   = (const float*)d_in[28]; const float* dbz  = (const float*)d_in[29];
  const float* dbr  = (const float*)d_in[30]; const float* dC   = (const float*)d_in[31];
  const float* dCz  = (const float*)d_in[32]; const float* dCr  = (const float*)d_in[33];
  const float* W_a  = (const float*)d_in[34]; const float* U_a  = (const float*)d_in[35];
  const float* v_a  = (const float*)d_in[36]; const float* U_o  = (const float*)d_in[37];
  const float* V_o  = (const float*)d_in[38]; const float* C_o  = (const float*)d_in[39];
  const float* W_o  = (const float*)d_in[40]; const float* W_s  = (const float*)d_in[41];
  float* out = (float*)d_out;

  // --- workspace bump allocator (deterministic; ~207 MB total) ---
  char* base = (char*)d_ws; size_t off = 0;
  auto alloc = [&](size_t bytes) -> void* {
    void* p = base + off; off = (off + bytes + 255) & ~(size_t)255; return p;
  };
  const int SB = SS * BB;                        // 3200 rows
  // bf16 weights
  bf16_t* wx3f = (bf16_t*)alloc((size_t)3 * H_PAD * E_PAD * 2);
  bf16_t* wx3b = (bf16_t*)alloc((size_t)3 * H_PAD * E_PAD * 2);
  bf16_t* wx3d = (bf16_t*)alloc((size_t)3 * H_PAD * E_PAD * 2);
  bf16_t* u2f  = (bf16_t*)alloc((size_t)2 * H_PAD * H_PAD * 2);
  bf16_t* u2b  = (bf16_t*)alloc((size_t)2 * H_PAD * H_PAD * 2);
  bf16_t* u2d  = (bf16_t*)alloc((size_t)2 * H_PAD * H_PAD * 2);
  bf16_t* ucf  = (bf16_t*)alloc((size_t)H_PAD * H_PAD * 2);
  bf16_t* ucb  = (bf16_t*)alloc((size_t)H_PAD * H_PAD * 2);
  bf16_t* ucd  = (bf16_t*)alloc((size_t)H_PAD * H_PAD * 2);
  bf16_t* cx3d = (bf16_t*)alloc((size_t)3 * H_PAD * C2PAD * 2);
  bf16_t* wa   = (bf16_t*)alloc((size_t)H_PAD * H_PAD * 2);
  bf16_t* ua   = (bf16_t*)alloc((size_t)H_PAD * C2PAD * 2);
  bf16_t* uo   = (bf16_t*)alloc((size_t)M2PAD * H_PAD * 2);
  bf16_t* vo   = (bf16_t*)alloc((size_t)M2PAD * E_PAD * 2);
  bf16_t* co   = (bf16_t*)alloc((size_t)M2PAD * C2PAD * 2);
  bf16_t* wo   = (bf16_t*)alloc((size_t)V_PAD * M_PAD * 2);
  bf16_t* wsm  = (bf16_t*)alloc((size_t)H_PAD * H_PAD * 2);
  float*  b3f  = (float*)alloc((size_t)3 * H_PAD * 4);
  float*  b3b  = (float*)alloc((size_t)3 * H_PAD * 4);
  float*  b3d  = (float*)alloc((size_t)3 * H_PAD * 4);
  float*  vad  = (float*)alloc((size_t)H_PAD * 4);
  // activations
  bf16_t* xsrc = (bf16_t*)alloc((size_t)SB * E_PAD * 2);     // [S,B,E] bf16
  bf16_t* yprv = (bf16_t*)alloc((size_t)SB * E_PAD * 2);     // [T,B,E] bf16
  bf16_t* X3f  = (bf16_t*)alloc((size_t)SB * 3 * H_PAD * 2);
  bf16_t* X3b  = (bf16_t*)alloc((size_t)SB * 3 * H_PAD * 2);
  bf16_t* X3d  = (bf16_t*)alloc((size_t)SB * 3 * H_PAD * 2);
  bf16_t* Yv   = (bf16_t*)alloc((size_t)SB * M2PAD * 2);
  bf16_t* hfb  = (bf16_t*)alloc((size_t)SB * H_PAD * 2);     // [S,B,H]
  bf16_t* hbb  = (bf16_t*)alloc((size_t)SB * H_PAD * 2);
  bf16_t* Hann = (bf16_t*)alloc((size_t)SB * C2PAD * 2);     // [B,S,C2PAD]
  float*  Uh   = (float*)alloc((size_t)SB * H_PAD * 4);
  float*  s_f  = (float*)alloc((size_t)BB * H_PAD * 4);
  bf16_t* s_bf = (bf16_t*)alloc((size_t)BB * H_PAD * 2);
  float*  zbuf = (float*)alloc((size_t)BB * H_PAD * 4);
  bf16_t* rs   = (bf16_t*)alloc((size_t)BB * H_PAD * 2);
  float*  G2   = (float*)alloc((size_t)BB * 2 * H_PAD * 4);
  float*  Gc   = (float*)alloc((size_t)BB * H_PAD * 4);
  float*  Gc3  = (float*)alloc((size_t)BB * 3 * H_PAD * 4);
  float*  Was  = (float*)alloc((size_t)BB * H_PAD * 4);
  float*  eng  = (float*)alloc((size_t)BB * SS * 4);
  float*  alp  = (float*)alloc((size_t)BB * SS * 4);
  bf16_t* cvec = (bf16_t*)alloc((size_t)BB * C2PAD * 2);
  float*  T1   = (float*)alloc((size_t)BB * M2PAD * 4);
  float*  T2   = (float*)alloc((size_t)BB * M2PAD * 4);
  bf16_t* tmx  = (bf16_t*)alloc((size_t)BB * M_PAD * 2);
  float*  Ws0  = (float*)alloc((size_t)BB * H_PAD * 4);

  // --- helpers ---
  auto gemm = [&](const bf16_t* A, const bf16_t* W, float* C, bf16_t* Cbf,
                  const float* bias, int M, int kdim, int lda, int ldw,
                  long long ldc, int n_pad, int n_valid) {
    dim3 g((n_pad + 255) / 256, M / 32);
    gemm_bf16_wmma<<<g, 256, 0, stream>>>(A, W, C, Cbf, bias, kdim, lda, ldw,
                                          ldc, n_pad, n_valid);
  };
  auto packw = [&](const float* src, bf16_t* dst, int rows_sec, int N, int K,
                   int K_pad, int split, int split_pad) {
    int blocks = (int)(((long long)rows_sec * K_pad) / 256);
    pack_w_kernel<<<blocks, 256, 0, stream>>>(src, dst, N, K, K_pad, split, split_pad);
  };
  auto packb = [&](const float* src, float* dst, int N, int sec) {
    pack_b_kernel<<<sec / 256, 256, 0, stream>>>(src, dst, N);
  };

  // --- 1. pack weights to zero-padded bf16 (one-time; L2-resident after) ---
  packw(efWr, wx3f + (size_t)0 * H_PAD * E_PAD, H_PAD, H_DIM, E_DIM, E_PAD, E_DIM, E_PAD);
  packw(efWz, wx3f + (size_t)1 * H_PAD * E_PAD, H_PAD, H_DIM, E_DIM, E_PAD, E_DIM, E_PAD);
  packw(efW,  wx3f + (size_t)2 * H_PAD * E_PAD, H_PAD, H_DIM, E_DIM, E_PAD, E_DIM, E_PAD);
  packw(ebWr, wx3b + (size_t)0 * H_PAD * E_PAD, H_PAD, H_DIM, E_DIM, E_PAD, E_DIM, E_PAD);
  packw(ebWz, wx3b + (size_t)1 * H_PAD * E_PAD, H_PAD, H_DIM, E_DIM, E_PAD, E_DIM, E_PAD);
  packw(ebW,  wx3b + (size_t)2 * H_PAD * E_PAD, H_PAD, H_DIM, E_DIM, E_PAD, E_DIM, E_PAD);
  packw(dWr,  wx3d + (size_t)0 * H_PAD * E_PAD, H_PAD, H_DIM, E_DIM, E_PAD, E_DIM, E_PAD);
  packw(dWz,  wx3d + (size_t)1 * H_PAD * E_PAD, H_PAD, H_DIM, E_DIM, E_PAD, E_DIM, E_PAD);
  packw(dW,   wx3d + (size_t)2 * H_PAD * E_PAD, H_PAD, H_DIM, E_DIM, E_PAD, E_DIM, E_PAD);
  packw(efUr, u2f + (size_t)0 * H_PAD * H_PAD, H_PAD, H_DIM, H_DIM, H_PAD, H_DIM, H_PAD);
  packw(efUz, u2f + (size_t)1 * H_PAD * H_PAD, H_PAD, H_DIM, H_DIM, H_PAD, H_DIM, H_PAD);
  packw(ebUr, u2b + (size_t)0 * H_PAD * H_PAD, H_PAD, H_DIM, H_DIM, H_PAD, H_DIM, H_PAD);
  packw(ebUz, u2b + (size_t)1 * H_PAD * H_PAD, H_PAD, H_DIM, H_DIM, H_PAD, H_DIM, H_PAD);
  packw(dUr,  u2d + (size_t)0 * H_PAD * H_PAD, H_PAD, H_DIM, H_DIM, H_PAD, H_DIM, H_PAD);
  packw(dUz,  u2d + (size_t)1 * H_PAD * H_PAD, H_PAD, H_DIM, H_DIM, H_PAD, H_DIM, H_PAD);
  packw(efU, ucf, H_PAD, H_DIM, H_DIM, H_PAD, H_DIM, H_PAD);
  packw(ebU, ucb, H_PAD, H_DIM, H_DIM, H_PAD, H_DIM, H_PAD);
  packw(dU,  ucd, H_PAD, H_DIM, H_DIM, H_PAD, H_DIM, H_PAD);
  // context-side weights use the split-K layout (hf @ [0,1000), hb @ [1024,2024))
  packw(dCr, cx3d + (size_t)0 * H_PAD * C2PAD, H_PAD, H_DIM, C2DIM, C2PAD, H_DIM, H_PAD);
  packw(dCz, cx3d + (size_t)1 * H_PAD * C2PAD, H_PAD, H_DIM, C2DIM, C2PAD, H_DIM, H_PAD);
  packw(dC,  cx3d + (size_t)2 * H_PAD * C2PAD, H_PAD, H_DIM, C2DIM, C2PAD, H_DIM, H_PAD);
  packw(W_a, wa, H_PAD, H_DIM, H_DIM, H_PAD, H_DIM, H_PAD);
  packw(U_a, ua, H_PAD, H_DIM, C2DIM, C2PAD, H_DIM, H_PAD);
  packw(U_o, uo, M2PAD, 2 * M_DIM, H_DIM, H_PAD, H_DIM, H_PAD);
  packw(V_o, vo, M2PAD, 2 * M_DIM, E_DIM, E_PAD, E_DIM, E_PAD);
  packw(C_o, co, M2PAD, 2 * M_DIM, C2DIM, C2PAD, H_DIM, H_PAD);
  packw(W_o, wo, V_PAD, V_DIM, M_DIM, M_PAD, M_DIM, M_PAD);
  packw(W_s, wsm, H_PAD, H_DIM, H_DIM, H_PAD, H_DIM, H_PAD);
  packb(efbr, b3f + 0 * H_PAD, H_DIM, H_PAD); packb(efbz, b3f + 1 * H_PAD, H_DIM, H_PAD);
  packb(efb,  b3f + 2 * H_PAD, H_DIM, H_PAD);
  packb(ebbr, b3b + 0 * H_PAD, H_DIM, H_PAD); packb(ebbz, b3b + 1 * H_PAD, H_DIM, H_PAD);
  packb(ebb,  b3b + 2 * H_PAD, H_DIM, H_PAD);
  packb(dbr,  b3d + 0 * H_PAD, H_DIM, H_PAD); packb(dbz,  b3d + 1 * H_PAD, H_DIM, H_PAD);
  packb(db,   b3d + 2 * H_PAD, H_DIM, H_PAD);
  packb(v_a,  vad, H_DIM, H_PAD);

  // --- 2. embeddings + batched x-projections (big WMMA GEMMs, M=3200) ---
  gather_src_kernel<<<(SB * E_PAD) / 256, 256, 0, stream>>>(src_emb, src_tok, xsrc);
  gather_tgt_kernel<<<(SB * E_PAD) / 256, 256, 0, stream>>>(tgt_emb, tgt_tok, yprv);
  gemm(xsrc, wx3f, nullptr, X3f, nullptr, SB, E_PAD, E_PAD, E_PAD, 3 * H_PAD, 3 * H_PAD, 3 * H_PAD);
  gemm(xsrc, wx3b, nullptr, X3b, nullptr, SB, E_PAD, E_PAD, E_PAD, 3 * H_PAD, 3 * H_PAD, 3 * H_PAD);
  gemm(yprv, wx3d, nullptr, X3d, nullptr, SB, E_PAD, E_PAD, E_PAD, 3 * H_PAD, 3 * H_PAD, 3 * H_PAD);
  gemm(yprv, vo,   nullptr, Yv,  nullptr, SB, E_PAD, E_PAD, E_PAD, M2PAD, M2PAD, M2PAD);

  const int NEH = (BB * H_PAD) / 256;          // 256 blocks

  // --- 3. encoder forward scan ---
  zero_f32_kernel <<<NEH, 256, 0, stream>>>(s_f);
  zero_bf16_kernel<<<NEH, 256, 0, stream>>>(s_bf);
  for (int t = 0; t < SS; ++t) {
    const bf16_t* x3 = X3f + (size_t)t * BB * 3 * H_PAD;
    gemm(s_bf, u2f, G2, nullptr, nullptr, BB, H_PAD, H_PAD, H_PAD, 2 * H_PAD, 2 * H_PAD, 2 * H_PAD);
    gru_rz_kernel<<<NEH, 256, 0, stream>>>(x3, G2, nullptr, s_f, b3f, rs, zbuf);
    gemm(rs, ucf, Gc, nullptr, nullptr, BB, H_PAD, H_PAD, H_PAD, H_PAD, H_PAD, H_PAD);
    gru_upd_kernel<<<NEH, 256, 0, stream>>>(x3, Gc, nullptr, zbuf, b3f, s_f, s_bf,
                                            hfb + (size_t)t * BB * H_PAD);
  }

  // --- 4. encoder backward scan ---
  zero_f32_kernel <<<NEH, 256, 0, stream>>>(s_f);
  zero_bf16_kernel<<<NEH, 256, 0, stream>>>(s_bf);
  for (int t = SS - 1; t >= 0; --t) {
    const bf16_t* x3 = X3b + (size_t)t * BB * 3 * H_PAD;
    gemm(s_bf, u2b, G2, nullptr, nullptr, BB, H_PAD, H_PAD, H_PAD, 2 * H_PAD, 2 * H_PAD, 2 * H_PAD);
    gru_rz_kernel<<<NEH, 256, 0, stream>>>(x3, G2, nullptr, s_f, b3b, rs, zbuf);
    gemm(rs, ucb, Gc, nullptr, nullptr, BB, H_PAD, H_PAD, H_PAD, H_PAD, H_PAD, H_PAD);
    gru_upd_kernel<<<NEH, 256, 0, stream>>>(x3, Gc, nullptr, zbuf, b3b, s_f, s_bf,
                                            hbb + (size_t)t * BB * H_PAD);
  }

  // --- 5. annotations, U_a h precompute, decoder s0 ---
  hann_kernel<<<((size_t)SB * C2PAD) / 256, 256, 0, stream>>>(hfb, hbb, Hann);
  gemm(Hann, ua, Uh, nullptr, nullptr, SB, C2PAD, C2PAD, C2PAD, H_PAD, H_PAD, H_PAD);
  gemm(s_bf, wsm, Ws0, nullptr, nullptr, BB, H_PAD, H_PAD, H_PAD, H_PAD, H_PAD, H_PAD); // s_bf == hb1
  tanh_s0_kernel<<<NEH, 256, 0, stream>>>(Ws0, s_f, s_bf);

  // --- 6. decoder scan with attention + maxout + logits ---
  for (int t = 0; t < SS; ++t) {
    // attention
    gemm(s_bf, wa, Was, nullptr, nullptr, BB, H_PAD, H_PAD, H_PAD, H_PAD, H_PAD, H_PAD);
    energy_kernel <<<BB * SS, 128, 0, stream>>>(Was, Uh, vad, eng);
    softmax_kernel<<<BB, 64, 0, stream>>>(eng, alp);
    context_kernel<<<(BB * C2PAD) / 256, 256, 0, stream>>>(alp, Hann, cvec);
    // output layer: t = U_o s + C_o c + V_o y  -> maxout -> W_o
    gemm(s_bf, uo, T1, nullptr, nullptr, BB, H_PAD, H_PAD, H_PAD, M2PAD, M2PAD, M2PAD);
    gemm(cvec, co, T2, nullptr, nullptr, BB, C2PAD, C2PAD, C2PAD, M2PAD, M2PAD, M2PAD);
    maxout_kernel<<<(BB * M_PAD) / 256, 256, 0, stream>>>(T1, T2,
                                                          Yv + (size_t)t * BB * M2PAD, tmx);
    gemm(tmx, wo, out + (size_t)t * V_DIM, nullptr, nullptr,
         BB, M_PAD, M_PAD, M_PAD, (long long)SS * V_DIM, V_PAD, V_DIM);
    // GRU state update (with context projections)
    const bf16_t* x3 = X3d + (size_t)t * BB * 3 * H_PAD;
    gemm(cvec, cx3d, Gc3, nullptr, nullptr, BB, C2PAD, C2PAD, C2PAD, 3 * H_PAD, 3 * H_PAD, 3 * H_PAD);
    gemm(s_bf, u2d, G2, nullptr, nullptr, BB, H_PAD, H_PAD, H_PAD, 2 * H_PAD, 2 * H_PAD, 2 * H_PAD);
    gru_rz_kernel<<<NEH, 256, 0, stream>>>(x3, G2, Gc3, s_f, b3d, rs, zbuf);
    gemm(rs, ucd, Gc, nullptr, nullptr, BB, H_PAD, H_PAD, H_PAD, H_PAD, H_PAD, H_PAD);
    gru_upd_kernel<<<NEH, 256, 0, stream>>>(x3, Gc, Gc3, zbuf, b3d, s_f, s_bf, nullptr);
  }
}
